// DySATNet_41764261987070
// MI455X (gfx1250) — compile-verified, matrix operand order checked by reading, not run on hardware
//
#include <hip/hip_runtime.h>
#include <hip/hip_bf16.h>
#include <math.h>

typedef __attribute__((ext_vector_type(16))) _Float16 v16h;
typedef __attribute__((ext_vector_type(8)))  float    v8f;

// ---------------------------------------------------------------------------
// B-packing: rearrange B(K,Nn) f32 into per-(kstep, ntile, lane) contiguous
// f16 WMMA fragments so each lane fetches its 16-half B fragment with one
// 32-byte contiguous load instead of 16 column-strided b32 loads.
// Fragment layout (ISA 16-bit B 32x16): lane half selects K base (0/8),
// element i -> k = base + (i<8 ? i : 8+i), col = lane&15.
// ---------------------------------------------------------------------------
__global__ void k_pack_b(const float* __restrict__ B, _Float16* __restrict__ Bp,
                         int K, int Nn)
{
  int tid = blockIdx.x * blockDim.x + threadIdx.x;
  if (tid >= K * Nn) return;
  const int nt   = Nn >> 4;
  const int i    = tid & 15;
  const int lane = (tid >> 4) & 31;
  const int blk  = tid >> 9;            // ks*nt + tn
  const int tn   = blk % nt;
  const int ks   = blk / nt;
  const int base = (lane & 16) ? 8 : 0;
  const int k    = ks * 32 + base + (i < 8 ? i : 8 + i);
  const int col  = tn * 16 + (lane & 15);
  Bp[tid] = (_Float16)B[(size_t)k * Nn + col];
}

// ---------------------------------------------------------------------------
// WMMA GEMM: C(M,Nn) = A(M,K) @ B(K,Nn) [+bias], f32 memory, f16 fragments,
// f32 accumulate. Each wave owns a 16(M) x 64(N) strip: one A fragment per
// K-step feeds 4 v_wmma_f32_16x16x32_f16 ops (register blocking along N).
// A fetched as float4 (global_load_b128) + v_cvt_pk_f16_f32; B fetched as a
// contiguous 32B pre-packed fragment.
// ---------------------------------------------------------------------------
__global__ void __launch_bounds__(256) k_gemm_wmma(
    const float*    __restrict__ A, int lda,
    const _Float16* __restrict__ Bp,
    const float*    __restrict__ bias,
    float*          __restrict__ C,
    int M, int K, int Nn)
{
  const int lane = threadIdx.x & 31;
  const int wave = threadIdx.x >> 5;
  const int nt = Nn >> 4;        // 16-col tiles
  const int gn = nt >> 2;        // 64-col groups (Nn is always a multiple of 64)
  const int mt = M >> 4;
  int strip = blockIdx.x * 8 + wave;
  if (strip >= mt * gn) return;  // uniform per wave -> EXEC all-ones below
  const int tm = strip / gn;
  const int tg = strip - tm * gn;
  const int half = lane >> 4;
  const int r    = lane & 15;
  const int ksteps = K >> 5;

  v8f acc0 = {}, acc1 = {}, acc2 = {}, acc3 = {};
  const float* Arow = A + (size_t)(tm * 16 + r) * lda + half * 8;
  const _Float16* Bbase = Bp + ((size_t)(tg * 4) * 32 + lane) * 16;

#pragma unroll 2
  for (int ks = 0; ks < ksteps; ++ks) {
    // ---- A fragment: 16 halves from 4x float4 contiguous loads ----
    const float* Ak = Arow + ks * 32;
    float4 f0 = *(const float4*)(Ak);
    float4 f1 = *(const float4*)(Ak + 4);
    float4 f2 = *(const float4*)(Ak + 16);
    float4 f3 = *(const float4*)(Ak + 20);
    v16h a;
    a[0]=(_Float16)f0.x; a[1]=(_Float16)f0.y; a[2]=(_Float16)f0.z; a[3]=(_Float16)f0.w;
    a[4]=(_Float16)f1.x; a[5]=(_Float16)f1.y; a[6]=(_Float16)f1.z; a[7]=(_Float16)f1.w;
    a[8]=(_Float16)f2.x; a[9]=(_Float16)f2.y; a[10]=(_Float16)f2.z; a[11]=(_Float16)f2.w;
    a[12]=(_Float16)f3.x; a[13]=(_Float16)f3.y; a[14]=(_Float16)f3.z; a[15]=(_Float16)f3.w;

    // ---- 4 pre-packed B fragments (32B contiguous each) ----
    const _Float16* Bk = Bbase + (size_t)ks * nt * 512;
    v16h b0 = *(const v16h*)(Bk);
    v16h b1 = *(const v16h*)(Bk + 512);
    v16h b2 = *(const v16h*)(Bk + 1024);
    v16h b3 = *(const v16h*)(Bk + 1536);

    acc0 = __builtin_amdgcn_wmma_f32_16x16x32_f16(false, a, false, b0, (short)0, acc0, false, false);
    acc1 = __builtin_amdgcn_wmma_f32_16x16x32_f16(false, a, false, b1, (short)0, acc1, false, false);
    acc2 = __builtin_amdgcn_wmma_f32_16x16x32_f16(false, a, false, b2, (short)0, acc2, false, false);
    acc3 = __builtin_amdgcn_wmma_f32_16x16x32_f16(false, a, false, b3, (short)0, acc3, false, false);
  }

  // ---- store: C/D layout row = 8*half + e, col = lane&15 ----
  v8f accs[4] = {acc0, acc1, acc2, acc3};
#pragma unroll
  for (int j = 0; j < 4; ++j) {
    const int col = (tg * 4 + j) * 16 + r;
    const float bb = bias ? bias[col] : 0.0f;
    float* Crow = C + (size_t)(tm * 16 + half * 8) * Nn + col;
#pragma unroll
    for (int e = 0; e < 8; ++e)
      Crow[(size_t)e * Nn] = accs[j][e] + bb;
  }
}

// ---------------------------------------------------------------------------
// GAT support kernels (bandwidth/atomic bound — VALU path)
// ---------------------------------------------------------------------------
__global__ void k_fill(float* __restrict__ p, float v, int n) {
  int tid = blockIdx.x * blockDim.x + threadIdx.x;
  if (tid < n) p[tid] = v;
}

// per (node, head): e_s = <hp[n,h,:], a_src[h,:]>, e_d likewise (64-wide dot)
__global__ void k_node_scores(const float* __restrict__ hp,
                              const float* __restrict__ asrc,
                              const float* __restrict__ adst,
                              float* __restrict__ es, float* __restrict__ ed,
                              int N)
{
  int tid = blockIdx.x * blockDim.x + threadIdx.x;
  if (tid >= N * 4) return;
  int n = tid >> 2, h = tid & 3;
  const float* p   = hp + (size_t)n * 256 + h * 64;
  const float* as_ = asrc + h * 64;
  const float* ad_ = adst + h * 64;
  float s = 0.f, d = 0.f;
#pragma unroll 8
  for (int f = 0; f < 64; ++f) { float v = p[f]; s += v * as_[f]; d += v * ad_[f]; }
  es[tid] = s; ed[tid] = d;
}

__device__ __forceinline__ void edge_sd(const int* ei, int E, int e, int& s, int& d) {
  if (e < E) { s = ei[e]; d = ei[E + e]; }     // real edge
  else       { s = e - E; d = s; }             // self loop
}

__global__ void k_edge_max(const int* __restrict__ ei, int E, int N,
                           const float* __restrict__ es, const float* __restrict__ ed,
                           float* __restrict__ mx)
{
  int tid = blockIdx.x * blockDim.x + threadIdx.x;
  int tot = (E + N) * 4;
  if (tid >= tot) return;
  int e = tid >> 2, h = tid & 3, s, d;
  edge_sd(ei, E, e, s, d);
  float v = es[s * 4 + h] + ed[d * 4 + h];
  v = v > 0.f ? v : 0.2f * v;                  // LeakyReLU(0.2)
  __hip_atomic_fetch_max(&mx[d * 4 + h], v, __ATOMIC_RELAXED,
                         __HIP_MEMORY_SCOPE_AGENT);
}

__global__ void k_edge_expsum(const int* __restrict__ ei, int E, int N,
                              const float* __restrict__ es, const float* __restrict__ ed,
                              const float* __restrict__ mx,
                              float* __restrict__ zs, float* __restrict__ wedge)
{
  int tid = blockIdx.x * blockDim.x + threadIdx.x;
  int tot = (E + N) * 4;
  if (tid >= tot) return;
  int e = tid >> 2, h = tid & 3, s, d;
  edge_sd(ei, E, e, s, d);
  float v = es[s * 4 + h] + ed[d * 4 + h];
  v = v > 0.f ? v : 0.2f * v;
  float w = __expf(v - mx[d * 4 + h]);
  wedge[tid] = w;
  atomicAdd(&zs[d * 4 + h], w);
}

__global__ void k_edge_agg(const int* __restrict__ ei, int E, int N,
                           const float* __restrict__ hp,
                           const float* __restrict__ wedge,
                           const float* __restrict__ zs,
                           float* __restrict__ agg)
{
  int tid = blockIdx.x * blockDim.x + threadIdx.x;
  int tot = (E + N) * 4;
  if (tid >= tot) return;
  int e = tid >> 2, h = tid & 3, s, d;
  edge_sd(ei, E, e, s, d);
  float alpha = wedge[tid] / (zs[d * 4 + h] + 1e-16f);
  const float* srcp = hp  + (size_t)s * 256 + h * 64;
  float*       dstp = agg + (size_t)d * 256 + h * 64;
#pragma unroll 4
  for (int f = 0; f < 64; ++f)
    atomicAdd(&dstp[f], srcp[f] * alpha);
}

// hout = ELU(agg + bias)
__global__ void k_gat_finalize(const float* __restrict__ agg,
                               const float* __restrict__ bias,
                               float* __restrict__ hout, int N)
{
  int tid = blockIdx.x * blockDim.x + threadIdx.x;
  if (tid >= N * 256) return;
  float v = agg[tid] + bias[tid & 255];
  hout[tid] = v > 0.f ? v : (__expf(v) - 1.0f);
}

// ---------------------------------------------------------------------------
// Temporal attention (collapsed algebraically: mean is linear -> Q/K from
// column means; only attn row T-1 is needed and sums to 1 -> V GEMM fuses
// into one weighted-combination GEMM).
// ---------------------------------------------------------------------------
__global__ void __launch_bounds__(256) k_colmean(const float* __restrict__ embs,
                                                 float* __restrict__ cm, int N)
{
  __shared__ float sh[256];
  int t = blockIdx.x >> 8, c = blockIdx.x & 255;
  const float* p = embs + (size_t)t * N * 256 + c;
  float s = 0.f;
  for (int n = threadIdx.x; n < N; n += 256) s += p[(size_t)n * 256];
  sh[threadIdx.x] = s;
  __syncthreads();
  for (int st = 128; st > 0; st >>= 1) {
    if (threadIdx.x < st) sh[threadIdx.x] += sh[threadIdx.x + st];
    __syncthreads();
  }
  if (threadIdx.x == 0) cm[t * 256 + c] = sh[0] / (float)N;
}

__global__ void __launch_bounds__(256) k_temporal_attn(
    const float* __restrict__ cm,
    const float* __restrict__ Wq, const float* __restrict__ bq,
    const float* __restrict__ Wk, const float* __restrict__ bk,
    float* __restrict__ attnw)
{
  __shared__ float sh[256];
  __shared__ float sc[4];
  const int c = threadIdx.x;
  float q = bq[c];
  const float* ml = cm + 3 * 256;
  for (int k = 0; k < 256; ++k) q += ml[k] * Wq[k * 256 + c];
  float kt[4];
  for (int t = 0; t < 4; ++t) {
    float v = bk[c];
    const float* mt = cm + t * 256;
    for (int k = 0; k < 256; ++k) v += mt[k] * Wk[k * 256 + c];
    kt[t] = v;
  }
  for (int t = 0; t < 4; ++t) {
    sh[c] = q * kt[t];
    __syncthreads();
    for (int st = 128; st > 0; st >>= 1) {
      if (c < st) sh[c] += sh[c + st];
      __syncthreads();
    }
    if (c == 0) sc[t] = sh[0] * (1.0f / 16.0f);   // /sqrt(256)
    __syncthreads();
  }
  if (c == 0) {
    float mxv = sc[0];
    for (int t = 1; t < 4; ++t) mxv = fmaxf(mxv, sc[t]);
    float z = 0.f, w[4];
    for (int t = 0; t < 4; ++t) { w[t] = __expf(sc[t] - mxv); z += w[t]; }
    for (int t = 0; t < 4; ++t) attnw[t] = w[t] / z;
  }
}

// combo = sum_t attnw[t] * embs[t]
__global__ void k_combine(const float* __restrict__ embs,
                          const float* __restrict__ attnw,
                          float* __restrict__ combo, int N)
{
  int tid = blockIdx.x * blockDim.x + threadIdx.x;
  if (tid >= N * 256) return;
  size_t stride = (size_t)N * 256;
  float v = attnw[0] * embs[tid]
          + attnw[1] * embs[stride + tid]
          + attnw[2] * embs[2 * stride + tid]
          + attnw[3] * embs[3 * stride + tid];
  combo[tid] = v;
}

// ---------------------------------------------------------------------------
extern "C" void kernel_launch(void* const* d_in, const int* in_sizes, int n_in,
                              void* d_out, int out_size, void* d_ws, size_t ws_size,
                              hipStream_t stream) {
  const float* x   = (const float*)d_in[0];
  const int*   ei  = (const int*)d_in[1];
  const float* Wp  = (const float*)d_in[2];
  const float* bp  = (const float*)d_in[3];
  const float* W0  = (const float*)d_in[4];
  const float* as0 = (const float*)d_in[5];
  const float* ad0 = (const float*)d_in[6];
  const float* b0  = (const float*)d_in[7];
  const float* W1  = (const float*)d_in[8];
  const float* as1 = (const float*)d_in[9];
  const float* ad1 = (const float*)d_in[10];
  const float* b1  = (const float*)d_in[11];
  const float* Wq  = (const float*)d_in[12];
  const float* bq  = (const float*)d_in[13];
  const float* Wk  = (const float*)d_in[14];
  const float* bk  = (const float*)d_in[15];
  const float* Wv  = (const float*)d_in[16];
  const float* bv  = (const float*)d_in[17];
  (void)n_in; (void)out_size; (void)ws_size;

  const int T = 4, Fin = 256, Hd = 64, heads = 4, D = 256;
  const int E = in_sizes[1] / 2;
  const int N = in_sizes[0] / (T * Fin);
  const int tot = E + N;

  float* f = (float*)d_ws;
  size_t o = 0;
  float* embs  = f + o; o += (size_t)4 * N * D;
  float* hproj = f + o; o += (size_t)N * Hd;
  float* hp    = f + o; o += (size_t)N * D;
  float* agg   = f + o; o += (size_t)N * D;
  float* h1    = f + o; o += (size_t)N * D;
  float* es    = f + o; o += (size_t)N * heads;
  float* ed    = f + o; o += (size_t)N * heads;
  float* mx    = f + o; o += (size_t)N * heads;
  float* zs    = f + o; o += (size_t)N * heads;
  float* wedge = f + o; o += (size_t)tot * heads;
  float* cm    = f + o; o += 4 * 256;
  float* attnw = f + o; o += 8;
  _Float16* Bp = (_Float16*)(f + o); o += (256 * 256) / 2 + 16;  // packed-B scratch (f16)
  float* combo = agg;   // safe reuse: agg of last GAT is dead by combine time

  auto gemm = [&](const float* A, int lda, const float* B, const float* bias,
                  float* C, int M, int K, int Nn) {
    int pk = K * Nn;
    k_pack_b<<<(pk + 255) / 256, 256, 0, stream>>>(B, Bp, K, Nn);
    int strips = (M >> 4) * (Nn >> 6);          // 16x64 strips per wave
    int blocks = (strips + 7) / 8;
    k_gemm_wmma<<<blocks, 256, 0, stream>>>(A, lda, Bp, bias, C, M, K, Nn);
  };

  auto gat = [&](const float* hin, int Kin, const float* W, const float* asrc,
                 const float* adst, const float* bias, float* hout) {
    gemm(hin, Kin, W, nullptr, hp, N, Kin, D);                     // hp = hin @ W
    k_node_scores<<<(N * heads + 255) / 256, 256, 0, stream>>>(hp, asrc, adst, es, ed, N);
    k_fill<<<(N * heads + 255) / 256, 256, 0, stream>>>(mx, -1e30f, N * heads);
    k_fill<<<(N * heads + 255) / 256, 256, 0, stream>>>(zs, 0.0f, N * heads);
    k_fill<<<(N * D + 255) / 256, 256, 0, stream>>>(agg, 0.0f, N * D);
    k_edge_max   <<<(tot * heads + 255) / 256, 256, 0, stream>>>(ei, E, N, es, ed, mx);
    k_edge_expsum<<<(tot * heads + 255) / 256, 256, 0, stream>>>(ei, E, N, es, ed, mx, zs, wedge);
    k_edge_agg   <<<(tot * heads + 255) / 256, 256, 0, stream>>>(ei, E, N, hp, wedge, zs, agg);
    k_gat_finalize<<<(N * D + 255) / 256, 256, 0, stream>>>(agg, bias, hout, N);
  };

  for (int t = 0; t < T; ++t) {
    gemm(x + (size_t)t * Fin, T * Fin, Wp, bp, hproj, N, Fin, Hd);  // h = x_t@Wp+bp
    gat(hproj, Hd, W0, as0, ad0, b0, h1);                           // ELU(GAT0)
    gat(h1,    D,  W1, as1, ad1, b1, embs + (size_t)t * N * D);     // ELU(GAT1)
  }

  k_colmean<<<dim3(4 * 256), dim3(256), 0, stream>>>(embs, cm, N);
  k_temporal_attn<<<1, 256, 0, stream>>>(cm, Wq, bq, Wk, bk, attnw);
  k_combine<<<(N * D + 255) / 256, 256, 0, stream>>>(embs, attnw, combo, N);
  gemm(combo, D, Wv, bv, (float*)d_out, N, D, D);                   // out = combo@Wv+bv
}
